// MLPBlock_70927089926420
// MI455X (gfx1250) — matrix operand order repaired and checked by reading, not compiled, not run on hardware
//
#include <hip/hip_runtime.h>
#include <hip/hip_bf16.h>

namespace {

constexpr int HD   = 2880;   // hidden
constexpr int ID   = 2880;   // intermediate
constexpr int EN   = 8;      // experts
constexpr int TOPK = 4;
constexpr int NT   = 1024;   // tokens
constexpr int BM   = 64;     // token rows per block (4 WMMA M-tiles)
constexpr int KC   = 64;     // K elements staged per round (2 WMMA K-steps)
constexpr int APAD = 72;     // LDS A row stride in halves (conflict-free b128)
constexpr float EPS_F   = 1e-5f;
constexpr float ALPHA_F = 1.702f;
constexpr float BETA_F  = 1.0f;

typedef __bf16 bf16_t;
typedef bf16_t v16bf __attribute__((ext_vector_type(16)));
typedef bf16_t v8bf  __attribute__((ext_vector_type(8)));
typedef bf16_t v4bf  __attribute__((ext_vector_type(4)));
typedef float  v8f_t __attribute__((ext_vector_type(8)));

// ---- workspace layout (bytes) ----
constexpr size_t OFF_TBF    = 0;                                   // N*H bf16
constexpr size_t SZ_TBF     = (size_t)NT * HD * 2;
constexpr size_t OFF_W4     = OFF_TBF + SZ_TBF;                    // N*4 f32
constexpr size_t SZ_W4      = (size_t)NT * TOPK * 4;
constexpr size_t OFF_CNT    = OFF_W4 + SZ_W4;                      // E ints
constexpr size_t SZ_CNT     = (size_t)EN * 4;
constexpr size_t OFF_LISTS  = OFF_CNT + SZ_CNT;                    // E*N ints
constexpr size_t SZ_LISTS   = (size_t)EN * NT * 4;
constexpr size_t OFF_ACT    = OFF_LISTS + SZ_LISTS;                // E*N*I bf16
constexpr size_t SZ_ACT     = (size_t)EN * NT * ID * 2;
constexpr size_t OFF_PART   = OFF_ACT + SZ_ACT;                    // N*4*H f32
// total ~100.3 MB

// ---------------- WMMA fragment helpers (CDNA5 layouts) ----------------

// A 16x32 bf16 fragment: lane m = lane&15, hi = lane>>4.
// halves 0..15 hold k = hi*16 + 0..15
__device__ inline v16bf load_a_frag(const bf16_t* As_k0, int hi) {
    v8bf lo = *(const v8bf*)(As_k0 + hi * 16);
    v8bf hh = *(const v8bf*)(As_k0 + hi * 16 + 8);
    v16bf a;
#pragma unroll
    for (int s = 0; s < 8; ++s) { a[s] = lo[s]; a[8 + s] = hh[s]; }
    return a;
}

// B 32x16 bf16 fragment: lane n = lane&15, hi = lane>>4; halves s hold
// k = hi*16 + s  -> 16 consecutive fp32 weights, converted to bf16.
__device__ inline v16bf load_b_frag_f32(const float* p) {
    v16bf b;
#pragma unroll
    for (int s = 0; s < 16; s += 4) {
        float4 f = *(const float4*)(p + s);
        b[s]     = (bf16_t)f.x;
        b[s + 1] = (bf16_t)f.y;
        b[s + 2] = (bf16_t)f.z;
        b[s + 3] = (bf16_t)f.w;
    }
    return b;
}

__device__ inline v8f_t zero8() {
    v8f_t z = {0.f, 0.f, 0.f, 0.f, 0.f, 0.f, 0.f, 0.f};
    return z;
}

// ---------------- Kernel 1: RMSNorm + router + expert list build ----------------

__global__ __launch_bounds__(256) void rmsnorm_router_kernel(
    const float* __restrict__ x, const float* __restrict__ norm_w,
    const float* __restrict__ gate_w, const float* __restrict__ gate_b,
    bf16_t* __restrict__ t_bf, float* __restrict__ w4,
    int* __restrict__ counts, int* __restrict__ lists)
{
    __shared__ float red[256];
    __shared__ float logits_s[EN];

    const int n   = blockIdx.x;
    const int tid = threadIdx.x;
    const float4* x4  = (const float4*)(x + (size_t)n * HD);
    const float4* nw4 = (const float4*)norm_w;

    float4 xs[3];
    int    vi[3];
    int    nv  = 0;
    float  ssq = 0.f;
    for (int v = tid; v < HD / 4; v += 256) {
        float4 a = x4[v];
        xs[nv] = a; vi[nv] = v; ++nv;
        ssq += a.x * a.x + a.y * a.y + a.z * a.z + a.w * a.w;
    }
    red[tid] = ssq;
    __syncthreads();
#pragma unroll
    for (int s = 128; s > 0; s >>= 1) {
        if (tid < s) red[tid] += red[tid + s];
        __syncthreads();
    }
    const float scale = rsqrtf(red[0] / (float)HD + EPS_F);
    __syncthreads();

    float pe[EN];
#pragma unroll
    for (int e = 0; e < EN; ++e) pe[e] = 0.f;

    for (int j = 0; j < nv; ++j) {
        const int v = vi[j];
        float4 w = nw4[v];
        float4 t;
        t.x = xs[j].x * scale * w.x;
        t.y = xs[j].y * scale * w.y;
        t.z = xs[j].z * scale * w.z;
        t.w = xs[j].w * scale * w.w;
        v4bf tb;
        tb[0] = (bf16_t)t.x; tb[1] = (bf16_t)t.y;
        tb[2] = (bf16_t)t.z; tb[3] = (bf16_t)t.w;
        *(v4bf*)(t_bf + (size_t)n * HD + (size_t)v * 4) = tb;
#pragma unroll
        for (int e = 0; e < EN; ++e) {
            float4 g = ((const float4*)(gate_w + (size_t)e * HD))[v];
            pe[e] += t.x * g.x + t.y * g.y + t.z * g.z + t.w * g.w;
        }
    }

    for (int e = 0; e < EN; ++e) {
        red[tid] = pe[e];
        __syncthreads();
#pragma unroll
        for (int s = 128; s > 0; s >>= 1) {
            if (tid < s) red[tid] += red[tid + s];
            __syncthreads();
        }
        if (tid == 0) logits_s[e] = red[0] + gate_b[e];
        __syncthreads();
    }

    if (tid == 0) {
        float l[EN];
        bool  used[EN];
#pragma unroll
        for (int e = 0; e < EN; ++e) { l[e] = logits_s[e]; used[e] = false; }
        int   idx[TOPK];
        float val[TOPK];
#pragma unroll
        for (int k = 0; k < TOPK; ++k) {
            int   bi = -1;
            float bv = -3.4e38f;
            for (int e = 0; e < EN; ++e) {
                if (!used[e] && l[e] > bv) { bv = l[e]; bi = e; }
            }
            used[bi] = true; idx[k] = bi; val[k] = bv;
        }
        const float m = val[0];
        float ex[TOPK], sum = 0.f;
#pragma unroll
        for (int k = 0; k < TOPK; ++k) { ex[k] = __expf(val[k] - m); sum += ex[k]; }
        const float inv = __builtin_amdgcn_rcpf(sum);
#pragma unroll
        for (int k = 0; k < TOPK; ++k) {
            const float wk  = ex[k] * inv;
            const int   e   = idx[k];
            const int   slot = atomicAdd(&counts[e], 1);
            lists[e * NT + slot] = n | (k << 16);
            w4[n * TOPK + k] = wk;
        }
    }
}

// ---------------- Kernel 2: gathered gate_up GEMM + SwiGLU -> act (bf16) ----------------
// grid: (ID/64, NT/BM, EN), block 128 (4 waves; each wave owns one 16-wide i tile,
// block owns a 64-token x 64-i output tile for both glu and lin halves)

__global__ __launch_bounds__(128) void gate_up_glu_kernel(
    const bf16_t* __restrict__ t_bf,
    const float* __restrict__ w_gu, const float* __restrict__ b_gu,
    const int* __restrict__ counts, const int* __restrict__ lists,
    bf16_t* __restrict__ act)
{
    const int e   = blockIdx.z;
    const int cnt = counts[e];
    const int s0  = blockIdx.y * BM;
    if (s0 >= cnt) return;
    const int i0  = blockIdx.x * 64;

    __shared__ bf16_t As[2][BM][APAD];   // double-buffered 64x64 bf16 A tile
    __shared__ int    toks[BM];

    const int tid = threadIdx.x;
    if (tid < BM) {
        int s = s0 + tid;
        if (s >= cnt) s = cnt - 1;
        toks[tid] = lists[e * NT + s] & 0xFFFF;
    }
    __syncthreads();

    const int lane = tid & 31;
    const int wv   = tid >> 5;           // wave 0..3
    const int l16  = lane & 15;          // A row (within tile) / B col
    const int hi   = lane >> 4;
    const int iw   = i0 + wv * 16;

    const int irow_g = iw + l16;
    const int irow_l = ID + iw + l16;
    const float* bg_base = w_gu + ((size_t)e * 2 * ID + irow_g) * HD;
    const float* bl_base = w_gu + ((size_t)e * 2 * ID + irow_l) * HD;

    // staging: 4 passes, each thread loads 8 halves of one row per pass
    const int rS = tid >> 3;             // 0..15 row within pass
    const int cS = (tid & 7) * 8;        // 0..56 col (halves)
    const bf16_t* gp[4];
#pragma unroll
    for (int p = 0; p < 4; ++p)
        gp[p] = t_bf + (size_t)toks[p * 16 + rS] * HD + cS;

    v8bf stage[4];
#pragma unroll
    for (int p = 0; p < 4; ++p) stage[p] = *(const v8bf*)(gp[p]);
#pragma unroll
    for (int p = 0; p < 4; ++p) *(v8bf*)&As[0][p * 16 + rS][cS] = stage[p];
    __syncthreads();

    v8f_t accg[4], accl[4];
#pragma unroll
    for (int t = 0; t < 4; ++t) { accg[t] = zero8(); accl[t] = zero8(); }

    constexpr int NR = HD / KC;          // 45 rounds
    for (int r = 0; r < NR; ++r) {
        const int cur  = r & 1;
        const bool more = (r + 1) < NR;
        if (more) {
            const int kn = (r + 1) * KC;
#pragma unroll
            for (int p = 0; p < 4; ++p) stage[p] = *(const v8bf*)(gp[p] + kn);
        }
#pragma unroll
        for (int kk = 0; kk < KC; kk += 32) {
            const int k0 = r * KC + kk;
            const v16bf bg = load_b_frag_f32(bg_base + k0 + hi * 16);
            const v16bf bl = load_b_frag_f32(bl_base + k0 + hi * 16);
#pragma unroll
            for (int t = 0; t < 4; ++t) {
                const v16bf a = load_a_frag(&As[cur][t * 16 + l16][kk], hi);
                accg[t] = __builtin_amdgcn_wmma_f32_16x16x32_bf16(false, a, false, bg, (short)0, accg[t], false, false);
                accl[t] = __builtin_amdgcn_wmma_f32_16x16x32_bf16(false, a, false, bl, (short)0, accl[t], false, false);
            }
        }
        if (more) {
#pragma unroll
            for (int p = 0; p < 4; ++p) *(v8bf*)&As[cur ^ 1][p * 16 + rS][cS] = stage[p];
        }
        __syncthreads();
    }

    const float bgv = b_gu[(size_t)e * 2 * ID + irow_g];
    const float blv = b_gu[(size_t)e * 2 * ID + irow_l];
#pragma unroll
    for (int t = 0; t < 4; ++t) {
#pragma unroll
        for (int r = 0; r < 8; ++r) {
            const int slot = s0 + t * 16 + r + hi * 8;
            const float g  = accg[t][r] + bgv;
            const float lv = accl[t][r] + blv;
            const float sig = __builtin_amdgcn_rcpf(1.0f + __expf(-ALPHA_F * g));
            const float av  = g * sig * (lv + BETA_F);
            act[((size_t)e * NT + slot) * ID + iw + l16] = (bf16_t)av;
        }
    }
}

// ---------------- Kernel 3: down GEMM, routing-weighted scatter -> part ----------------
// grid: (HD/64, NT/BM, EN), block 128

__global__ __launch_bounds__(128) void down_kernel(
    const bf16_t* __restrict__ act,
    const float* __restrict__ w_down, const float* __restrict__ b_down,
    const int* __restrict__ counts, const int* __restrict__ lists,
    const float* __restrict__ w4, float* __restrict__ part)
{
    const int e   = blockIdx.z;
    const int cnt = counts[e];
    const int s0  = blockIdx.y * BM;
    if (s0 >= cnt) return;
    const int h0  = blockIdx.x * 64;

    __shared__ bf16_t As[2][BM][APAD];
    __shared__ int    info_s[BM];
    __shared__ float  wgt_s[BM];

    const int tid = threadIdx.x;
    if (tid < BM) {
        int s = s0 + tid;
        int sc = s < cnt ? s : cnt - 1;
        const int packed = lists[e * NT + sc];
        info_s[tid] = packed;
        wgt_s[tid]  = w4[(packed & 0xFFFF) * TOPK + (packed >> 16)];
    }
    __syncthreads();

    const int lane = tid & 31;
    const int wv   = tid >> 5;
    const int l16  = lane & 15;
    const int hi   = lane >> 4;
    const int hcol = h0 + wv * 16 + l16;

    const float* bbase = w_down + ((size_t)e * HD + hcol) * ID;

    const int rS = tid >> 3;
    const int cS = (tid & 7) * 8;
    const bf16_t* ap[4];
#pragma unroll
    for (int p = 0; p < 4; ++p)
        ap[p] = act + ((size_t)e * NT + (s0 + p * 16 + rS)) * ID + cS;

    v8bf stage[4];
#pragma unroll
    for (int p = 0; p < 4; ++p) stage[p] = *(const v8bf*)(ap[p]);
#pragma unroll
    for (int p = 0; p < 4; ++p) *(v8bf*)&As[0][p * 16 + rS][cS] = stage[p];
    __syncthreads();

    v8f_t acc[4];
#pragma unroll
    for (int t = 0; t < 4; ++t) acc[t] = zero8();

    constexpr int NR = ID / KC;          // 45 rounds
    for (int r = 0; r < NR; ++r) {
        const int cur  = r & 1;
        const bool more = (r + 1) < NR;
        if (more) {
            const int kn = (r + 1) * KC;
#pragma unroll
            for (int p = 0; p < 4; ++p) stage[p] = *(const v8bf*)(ap[p] + kn);
        }
#pragma unroll
        for (int kk = 0; kk < KC; kk += 32) {
            const int k0 = r * KC + kk;
            const v16bf b = load_b_frag_f32(bbase + k0 + hi * 16);
#pragma unroll
            for (int t = 0; t < 4; ++t) {
                const v16bf a = load_a_frag(&As[cur][t * 16 + l16][kk], hi);
                acc[t] = __builtin_amdgcn_wmma_f32_16x16x32_bf16(false, a, false, b, (short)0, acc[t], false, false);
            }
        }
        if (more) {
#pragma unroll
            for (int p = 0; p < 4; ++p) *(v8bf*)&As[cur ^ 1][p * 16 + rS][cS] = stage[p];
        }
        __syncthreads();
    }

    const float bd = b_down[(size_t)e * HD + hcol];
#pragma unroll
    for (int t = 0; t < 4; ++t) {
#pragma unroll
        for (int r = 0; r < 8; ++r) {
            const int m = t * 16 + r + hi * 8;
            if (s0 + m < cnt) {
                const int packed = info_s[m];
                const int n = packed & 0xFFFF;
                const int k = packed >> 16;
                part[((size_t)(n * TOPK + k)) * HD + hcol] = wgt_s[m] * (acc[t][r] + bd);
            }
        }
    }
}

// ---------------- Kernel 4: out = x + sum_k part[n][k][:] (fixed order) ----------------

__global__ __launch_bounds__(256) void finalize_kernel(
    const float* __restrict__ x, const float* __restrict__ part,
    float* __restrict__ out)
{
    const size_t idx4   = (size_t)blockIdx.x * 256 + threadIdx.x;  // float4 index
    const size_t total4 = (size_t)NT * HD / 4;
    if (idx4 >= total4) return;
    const size_t n  = idx4 / (HD / 4);
    const size_t h4 = idx4 % (HD / 4);

    float4 acc = ((const float4*)x)[idx4];
    const float4* p = (const float4*)(part + n * TOPK * HD) + h4;
#pragma unroll
    for (int k = 0; k < TOPK; ++k) {
        float4 v = p[(size_t)k * (HD / 4)];
        acc.x += v.x; acc.y += v.y; acc.z += v.z; acc.w += v.w;
    }
    ((float4*)out)[idx4] = acc;
}

}  // namespace

extern "C" void kernel_launch(void* const* d_in, const int* in_sizes, int n_in,
                              void* d_out, int out_size, void* d_ws, size_t ws_size,
                              hipStream_t stream) {
    const float* x      = (const float*)d_in[0];
    const float* norm_w = (const float*)d_in[1];
    const float* gate_w = (const float*)d_in[2];
    const float* gate_b = (const float*)d_in[3];
    const float* w_gu   = (const float*)d_in[4];
    const float* b_gu   = (const float*)d_in[5];
    const float* w_dn   = (const float*)d_in[6];
    const float* b_dn   = (const float*)d_in[7];

    char* ws = (char*)d_ws;
    bf16_t* t_bf  = (bf16_t*)(ws + OFF_TBF);
    float*  w4    = (float*) (ws + OFF_W4);
    int*    cnts  = (int*)   (ws + OFF_CNT);
    int*    lists = (int*)   (ws + OFF_LISTS);
    bf16_t* act   = (bf16_t*)(ws + OFF_ACT);
    float*  part  = (float*) (ws + OFF_PART);

    hipMemsetAsync(cnts, 0, EN * sizeof(int), stream);

    rmsnorm_router_kernel<<<NT, 256, 0, stream>>>(
        x, norm_w, gate_w, gate_b, t_bf, w4, cnts, lists);

    gate_up_glu_kernel<<<dim3(ID / 64, NT / BM, EN), 128, 0, stream>>>(
        t_bf, w_gu, b_gu, cnts, lists, act);

    down_kernel<<<dim3(HD / 64, NT / BM, EN), 128, 0, stream>>>(
        act, w_dn, b_dn, cnts, lists, w4, part);

    finalize_kernel<<<(NT * HD / 4 + 255) / 256, 256, 0, stream>>>(
        x, part, (float*)d_out);
}
